// EncoderDecoder_33964601377474
// MI455X (gfx1250) — matrix operand order repaired and checked by reading, not compiled, not run on hardware
//
#include <hip/hip_runtime.h>
#include <hip/hip_bf16.h>
#include <math.h>

// ---------------------------------------------------------------------------
// MI455X (gfx1250) GRU seq2seq + attention + copy model.
// All GEMMs use v_wmma_f32_16x16x32_f16 (f16 in, f32 accumulate), B=8 padded
// to M=16. Sequential paths (encoder scan, decoder step) run as persistent
// single-workgroup WMMA kernels with LDS-resident state; only the vocab
// projection (N=32000) is a device-wide GEMM per decoder step.
// ---------------------------------------------------------------------------

#define BB 8
#define SS 400
#define TT 100
#define EE 256
#define HH 512
#define AA 256
#define VV 32000

typedef __attribute__((ext_vector_type(16))) _Float16 v16h;
typedef __attribute__((ext_vector_type(8)))  _Float16 v8h;
typedef __attribute__((ext_vector_type(8)))  float    v8f;

// ------------------------------ weight packing ------------------------------
// W is [N][K] row-major f32 (computing C = A @ W^T, so WMMA B[k][n] = W[n][k]).
// Packed tile (tn,tk): 512 halves at ((tn*Ktiles + tk)*512); element (lane,i)
// = W[tn*16 + (lane&15)][tk*32 + (lane<16?0:8) + (i<8 ? i : 8 + i)]   (ISA 16-bit B layout).
__global__ void pack_b(const float* __restrict__ W, _Float16* __restrict__ P,
                       int N, int K) {
  size_t idx = (size_t)blockIdx.x * blockDim.x + threadIdx.x;
  size_t total = (size_t)N * K;
  if (idx >= total) return;
  int Ktiles = K >> 5;
  size_t tile = idx >> 9;
  int within = (int)(idx & 511);
  int lane = within >> 4;
  int i = within & 15;
  int tn = (int)(tile / Ktiles);
  int tk = (int)(tile % Ktiles);
  int n = tn * 16 + (lane & 15);
  int k = tk * 32 + ((lane < 16) ? 0 : 8) + ((i < 8) ? i : 8 + i);
  P[idx] = (_Float16)W[(size_t)n * K + k];
}

// ------------------------------ generic WMMA GEMM ---------------------------
// C[Mtiles*16][ldc] (f32) = A[Mtiles*16][lda] (f16, row-major) @ Bpacked^T
// grid = (N/64, Mtiles), block = 128 (4 waves), one 16x16 tile per wave.
__global__ __launch_bounds__(128) void gemm_wmma(
    const _Float16* __restrict__ A, int lda,
    const _Float16* __restrict__ Bp,
    float* __restrict__ C, int ldc, int N, int K) {
  const int wave = threadIdx.x >> 5;
  const int lane = threadIdx.x & 31;
  const int tn = blockIdx.x * 4 + wave;
  if (tn * 16 >= N) return;                 // wave-uniform: EXEC stays all-1s
  const int tm = blockIdx.y;
  const int Ktiles = K >> 5;
  const int laneM = lane & 15;
  const int kb = (lane < 16) ? 0 : 8;
  const _Float16* Arow = A + (size_t)(tm * 16 + laneM) * lda + kb;
  const _Float16* Bt = Bp + (size_t)tn * Ktiles * 512 + lane * 16;
  v8f acc = {};
  for (int kt = 0; kt < Ktiles; ++kt) {
    v8h a0 = *(const v8h*)(Arow + kt * 32);
    v8h a1 = *(const v8h*)(Arow + kt * 32 + 16);
    v16h a = __builtin_shufflevector(a0, a1, 0,1,2,3,4,5,6,7,8,9,10,11,12,13,14,15);
    v16h b = *(const v16h*)(Bt + (size_t)kt * 512);
    if (kt + 1 < Ktiles)
      __builtin_prefetch(Bt + (size_t)(kt + 1) * 512, 0, 3); // global_prefetch
    acc = __builtin_amdgcn_wmma_f32_16x16x32_f16(false, a, false, b,
                                                 (short)0, acc, false, false);
  }
  const int m0 = tm * 16 + ((lane < 16) ? 0 : 8);
  const int n = tn * 16 + laneM;
#pragma unroll
  for (int r = 0; r < 8; ++r) C[(size_t)(m0 + r) * ldc + n] = acc[r];
}

// ------------------------------ embedding gather ----------------------------
// xs_f16[row = t*8 + b][e] = embed[src_[b][t]][e]
__global__ void embed_src_k(const float* __restrict__ embed,
                            const int* __restrict__ srci,
                            _Float16* __restrict__ xsf16) {
  size_t i = (size_t)blockIdx.x * blockDim.x + threadIdx.x;
  if (i >= (size_t)SS * BB * EE) return;
  size_t row = i >> 8;
  int e = (int)(i & 255);
  int t = (int)(row / BB), b = (int)(row % BB);
  int tok = srci[b * SS + t];
  xsf16[i] = (_Float16)embed[(size_t)tok * EE + e];
}

// ------------------------------ encoder scan --------------------------------
// One persistent workgroup per direction (grid=2). h kept in LDS; per step:
// gh = h @ Whh^T via WMMA (96 N-tiles over 32 waves), then GRU elementwise.
__global__ __launch_bounds__(1024) void enc_scan(
    const float* __restrict__ giF, const float* __restrict__ giB,
    const _Float16* __restrict__ WhhPF, const _Float16* __restrict__ WhhPB,
    const float* __restrict__ bihF, const float* __restrict__ bhhF,
    const float* __restrict__ bihB, const float* __restrict__ bhhB,
    float* __restrict__ ehid32, _Float16* __restrict__ ehid16,
    float* __restrict__ encfin) {
  extern __shared__ __align__(16) char smem[];
  _Float16* hF = (_Float16*)smem;                       // 16x512 f16
  float* ghS = (float*)(smem + 16 * 512 * 2);           // 8x1536 f32
  float* hS  = ghS + 8 * 1536;                          // 8x512  f32
  const int dir = blockIdx.x;
  const float* gi_all = dir ? giB : giF;
  const _Float16* WhhP = dir ? WhhPB : WhhPF;
  const float* bih = dir ? bihB : bihF;
  const float* bhh = dir ? bhhB : bhhF;
  const int tid = threadIdx.x;
  for (int i = tid; i < 16 * 512; i += 1024) hF[i] = (_Float16)0.0f;
  for (int i = tid; i < 8 * 512; i += 1024) hS[i] = 0.0f;
  __syncthreads();
  const int wave = tid >> 5, lane = tid & 31;
  const int laneM = lane & 15;
  const int kb = (lane < 16) ? 0 : 8;
  const int m0 = (lane < 16) ? 0 : 8;
  for (int step = 0; step < SS; ++step) {
    const int t = dir ? (SS - 1 - step) : step;
    // warm WGP caches with next step's gi slice (48KB) during the GEMM
    if (step + 1 < SS) {
      const int t2 = dir ? (SS - 2 - step) : (step + 1);
      const float* gnext = gi_all + (size_t)t2 * BB * 1536;
      if (tid < 384) __builtin_prefetch(gnext + tid * 32, 0, 3);
    }
    // gh = h @ Whh^T  (N=1536, K=512)
    for (int tt = 0; tt < 3; ++tt) {
      const int tn = wave * 3 + tt;
      const _Float16* Bt = WhhP + (size_t)tn * 16 * 512 + lane * 16;
      const _Float16* Arow = hF + laneM * 512 + kb;
      v8f acc = {};
#pragma unroll
      for (int kt = 0; kt < 16; ++kt) {
        v8h a0 = *(const v8h*)(Arow + kt * 32);
        v8h a1 = *(const v8h*)(Arow + kt * 32 + 16);
        v16h a = __builtin_shufflevector(a0, a1, 0,1,2,3,4,5,6,7,8,9,10,11,12,13,14,15);
        v16h b = *(const v16h*)(Bt + (size_t)kt * 512);
        acc = __builtin_amdgcn_wmma_f32_16x16x32_f16(false, a, false, b,
                                                     (short)0, acc, false, false);
      }
      if (m0 < 8) {                                     // only real batch rows
#pragma unroll
        for (int r = 0; r < 8; ++r)
          ghS[(m0 + r) * 1536 + tn * 16 + laneM] = acc[r];
      }
    }
    __syncthreads();
    const float* gi = gi_all + (size_t)t * BB * 1536;
    for (int idx = tid; idx < BB * HH; idx += 1024) {
      const int b = idx >> 9, j = idx & 511;
      float ir = gi[b * 1536 + j]        + bih[j];
      float iz = gi[b * 1536 + 512 + j]  + bih[512 + j];
      float in_ = gi[b * 1536 + 1024 + j] + bih[1024 + j];
      float hr = ghS[b * 1536 + j]        + bhh[j];
      float hz = ghS[b * 1536 + 512 + j]  + bhh[512 + j];
      float hn = ghS[b * 1536 + 1024 + j] + bhh[1024 + j];
      float r = 1.0f / (1.0f + __expf(-(ir + hr)));
      float z = 1.0f / (1.0f + __expf(-(iz + hz)));
      float n = tanhf(in_ + r * hn);
      float hnew = (1.0f - z) * n + z * hS[idx];
      hS[idx] = hnew;
      hF[b * 512 + j] = (_Float16)hnew;
      size_t o = ((size_t)b * SS + t) * 1024 + dir * 512 + j;
      ehid32[o] = hnew;
      ehid16[o] = (_Float16)hnew;
    }
    __syncthreads();
  }
  for (int idx = tid; idx < BB * HH; idx += 1024) {
    const int b = idx >> 9, j = idx & 511;
    encfin[b * 1024 + dir * 512 + j] = hS[idx];
  }
}

// ------------------------------ attribute init ------------------------------
__global__ void init_attrs(
    const float* __restrict__ uembed, const float* __restrict__ pembed,
    const int* __restrict__ user, const int* __restrict__ product,
    const float* __restrict__ mixW, const float* __restrict__ mixb,
    const float* __restrict__ encattrW, const float* __restrict__ encattrb,
    const float* __restrict__ initW, const float* __restrict__ initb,
    const float* __restrict__ attrKey, const float* __restrict__ encfin,
    float* __restrict__ encattr_out, float* __restrict__ pka,
    float* __restrict__ h, _Float16* __restrict__ xf16) {
  __shared__ float upm[768];
  __shared__ float afin[512];
  const int b = blockIdx.x, tid = threadIdx.x;   // block = 256
  const float* u = uembed + (size_t)user[b] * AA;
  const float* p = pembed + (size_t)product[b] * AA;
  for (int j = tid; j < 256; j += 256) { upm[j] = u[j]; upm[256 + j] = p[j]; }
  for (int j = tid; j < 256; j += 256) {
    float s = mixb[j];
    for (int k = 0; k < 256; ++k) s += u[k] * mixW[j * 512 + k];
    for (int k = 0; k < 256; ++k) s += p[k] * mixW[j * 512 + 256 + k];
    upm[512 + j] = (s > 0.0f) ? s : 0.01f * s;       // leaky_relu
  }
  __syncthreads();
  for (int j = tid; j < 512; j += 256) {
    float s = encattrb[j];
    for (int k = 0; k < 768; ++k) s += upm[k] * encattrW[j * 768 + k];
    afin[j] = (s > 0.0f) ? s : 0.01f * s;
  }
  for (int j = tid; j < 768; j += 256) encattr_out[b * 768 + j] = upm[j];
  __syncthreads();
  for (int j = tid; j < 3 * 512; j += 256) {
    int kk = j / 512, hh = j % 512;
    float s = 0.0f;
    for (int a = 0; a < 256; ++a) s += attrKey[hh * 256 + a] * upm[kk * 256 + a];
    pka[b * 1536 + j] = s;
  }
  for (int j = tid; j < 512; j += 256) {
    float s = initb[j];
    const float* ef = encfin + b * 1024;
    for (int k = 0; k < 1024; ++k) s += ef[k] * initW[j * 1536 + k];
    for (int k = 0; k < 512; ++k) s += afin[k] * initW[j * 1536 + 1024 + k];
    float v = (s > 0.0f) ? s : 0.01f * s;
    h[b * 512 + j] = v;
    xf16[b * 768 + 256 + j] = (_Float16)v;           // ch0 = h_dec
  }
}

// zero padding rows 8..15 of the M=16 f16 staging buffers (once per launch)
__global__ void zero_pads(_Float16* __restrict__ xf16,
                          _Float16* __restrict__ actx) {
  int i = blockIdx.x * blockDim.x + threadIdx.x;
  if (i < 8 * 768) xf16[8 * 768 + i] = (_Float16)0.0f;
  if (i < 8 * 1792) actx[8 * 1792 + i] = (_Float16)0.0f;
}

// ------------------------- fused decoder GRU step ---------------------------
// One 32-wave workgroup: build x=[pe,ch] and h (f16) in LDS, WMMA both GRU
// GEMMs (gi: K=768, gh: K=512; 96 N-tiles each), stage in LDS, elementwise.
__global__ __launch_bounds__(1024) void dec_fused_gru(
    const float* __restrict__ embed, const int* __restrict__ trg,
    const _Float16* __restrict__ xf16g,     // ch columns valid (prev step)
    const _Float16* __restrict__ WihP, const _Float16* __restrict__ WhhP,
    const float* __restrict__ bih, const float* __restrict__ bhh,
    float* __restrict__ h, int t) {
  extern __shared__ __align__(16) char smem[];
  _Float16* xL = (_Float16*)smem;                 // 16x768 f16
  _Float16* hL = xL + 16 * 768;                   // 16x512 f16
  float* giS = (float*)(hL + 16 * 512);           // 8x1536 f32
  float* ghS = giS + 8 * 1536;                    // 8x1536 f32
  const int tid = threadIdx.x;
  for (int i = tid; i < 16 * 768; i += 1024) {
    int r = i / 768, c = i - r * 768;
    _Float16 v = (_Float16)0.0f;
    if (r < 8) {
      if (c < 256) {
        int tok = (t == 0) ? 1 : trg[r * TT + (t - 1)];
        v = (_Float16)embed[(size_t)tok * EE + c];
      } else {
        v = xf16g[r * 768 + c];
      }
    }
    xL[i] = v;
  }
  for (int i = tid; i < 16 * 512; i += 1024) {
    int r = i >> 9, c = i & 511;
    hL[i] = (r < 8) ? (_Float16)h[r * 512 + c] : (_Float16)0.0f;
  }
  __syncthreads();
  const int wave = tid >> 5, lane = tid & 31, laneM = lane & 15;
  const int kb = (lane < 16) ? 0 : 8;
  const int m0 = (lane < 16) ? 0 : 8;
  for (int tt = 0; tt < 3; ++tt) {
    const int tn = wave * 3 + tt;
    {   // gi tile: x @ Wih^T, K=768
      const _Float16* Bt = WihP + (size_t)tn * 24 * 512 + lane * 16;
      const _Float16* Arow = xL + laneM * 768 + kb;
      v8f acc = {};
#pragma unroll
      for (int kt = 0; kt < 24; ++kt) {
        v8h a0 = *(const v8h*)(Arow + kt * 32);
        v8h a1 = *(const v8h*)(Arow + kt * 32 + 16);
        v16h a = __builtin_shufflevector(a0, a1, 0,1,2,3,4,5,6,7,8,9,10,11,12,13,14,15);
        v16h b = *(const v16h*)(Bt + (size_t)kt * 512);
        acc = __builtin_amdgcn_wmma_f32_16x16x32_f16(false, a, false, b,
                                                     (short)0, acc, false, false);
      }
      if (m0 < 8) {
#pragma unroll
        for (int r = 0; r < 8; ++r)
          giS[(m0 + r) * 1536 + tn * 16 + laneM] = acc[r];
      }
    }
    {   // gh tile: h @ Whh^T, K=512
      const _Float16* Bt = WhhP + (size_t)tn * 16 * 512 + lane * 16;
      const _Float16* Arow = hL + laneM * 512 + kb;
      v8f acc = {};
#pragma unroll
      for (int kt = 0; kt < 16; ++kt) {
        v8h a0 = *(const v8h*)(Arow + kt * 32);
        v8h a1 = *(const v8h*)(Arow + kt * 32 + 16);
        v16h a = __builtin_shufflevector(a0, a1, 0,1,2,3,4,5,6,7,8,9,10,11,12,13,14,15);
        v16h b = *(const v16h*)(Bt + (size_t)kt * 512);
        acc = __builtin_amdgcn_wmma_f32_16x16x32_f16(false, a, false, b,
                                                     (short)0, acc, false, false);
      }
      if (m0 < 8) {
#pragma unroll
        for (int r = 0; r < 8; ++r)
          ghS[(m0 + r) * 1536 + tn * 16 + laneM] = acc[r];
      }
    }
  }
  __syncthreads();
  for (int idx = tid; idx < BB * HH; idx += 1024) {
    const int b = idx >> 9, j = idx & 511;
    float ir = giS[b * 1536 + j]        + bih[j];
    float iz = giS[b * 1536 + 512 + j]  + bih[512 + j];
    float in_ = giS[b * 1536 + 1024 + j] + bih[1024 + j];
    float hr = ghS[b * 1536 + j]        + bhh[j];
    float hz = ghS[b * 1536 + 512 + j]  + bhh[512 + j];
    float hn = ghS[b * 1536 + 1024 + j] + bhh[1024 + j];
    float r = 1.0f / (1.0f + __expf(-(ir + hr)));
    float z = 1.0f / (1.0f + __expf(-(iz + hz)));
    float n = tanhf(in_ + r * hn);
    h[idx] = (1.0f - z) * n + z * h[idx];
  }
}

// attention over source + attributes; builds ctxhid A row and gp (grid = B)
__global__ void dec_attn(
    const float* __restrict__ h, const float* __restrict__ projkey,
    const float* __restrict__ ehid32,
    const float* __restrict__ attQ, const float* __restrict__ attE,
    const float* __restrict__ attrQ, const float* __restrict__ attrE,
    const float* __restrict__ pka, const float* __restrict__ encattr,
    const int* __restrict__ srcmask,
    const float* __restrict__ genpW, const float* __restrict__ genpb,
    const float* __restrict__ embed, const int* __restrict__ trg,
    float* __restrict__ alphas, _Float16* __restrict__ actx,
    float* __restrict__ gp, int t) {
  __shared__ float qp[512];
  __shared__ float qa[512];
  __shared__ float sc[SS];
  __shared__ float red[256];
  __shared__ float ctx[1024];
  __shared__ float ctxa[256];
  const int b = blockIdx.x, tid = threadIdx.x;   // block = 256
  const float* q = h + b * 512;
  for (int j = tid; j < 512; j += 256) {
    float s0 = 0.0f, s1 = 0.0f;
    for (int k = 0; k < 512; ++k) {
      s0 += q[k] * attQ[j * 512 + k];
      s1 += q[k] * attrQ[j * 512 + k];
    }
    qp[j] = s0; qa[j] = s1;
  }
  __syncthreads();
  for (int s = tid; s < SS; s += 256) {
    const float* pk = projkey + ((size_t)b * SS + s) * 512;
    float e = 0.0f;
    for (int hh = 0; hh < 512; ++hh) e += attE[hh] * tanhf(qp[hh] + pk[hh]);
    sc[s] = (srcmask[b * SS + s] == 0) ? -__builtin_inff() : e;
  }
  __syncthreads();
  float m = -__builtin_inff();
  for (int s = tid; s < SS; s += 256) m = fmaxf(m, sc[s]);
  red[tid] = m; __syncthreads();
  for (int o = 128; o > 0; o >>= 1) {
    if (tid < o) red[tid] = fmaxf(red[tid], red[tid + o]);
    __syncthreads();
  }
  m = red[0]; __syncthreads();
  float sum = 0.0f;
  for (int s = tid; s < SS; s += 256) { float e = __expf(sc[s] - m); sc[s] = e; sum += e; }
  red[tid] = sum; __syncthreads();
  for (int o = 128; o > 0; o >>= 1) {
    if (tid < o) red[tid] += red[tid + o];
    __syncthreads();
  }
  float inv = 1.0f / red[0]; __syncthreads();
  for (int s = tid; s < SS; s += 256) {
    float a = sc[s] * inv; sc[s] = a; alphas[b * SS + s] = a;
  }
  __syncthreads();
  for (int d = tid; d < 1024; d += 256) {
    float c = 0.0f;
    for (int s = 0; s < SS; ++s) c += sc[s] * ehid32[((size_t)b * SS + s) * 1024 + d];
    ctx[d] = c;
  }
  __syncthreads();
  if (tid < 3) {
    const float* pk = pka + (b * 3 + tid) * 512;
    float e = 0.0f;
    for (int hh = 0; hh < 512; ++hh) e += attrE[hh] * tanhf(qa[hh] + pk[hh]);
    red[tid] = e;
  }
  __syncthreads();
  if (tid == 0) {
    float m3 = fmaxf(red[0], fmaxf(red[1], red[2]));
    float e0 = __expf(red[0] - m3), e1 = __expf(red[1] - m3), e2 = __expf(red[2] - m3);
    float is = 1.0f / (e0 + e1 + e2);
    red[0] = e0 * is; red[1] = e1 * is; red[2] = e2 * is;
  }
  __syncthreads();
  float a0 = red[0], a1 = red[1], a2 = red[2];
  __syncthreads();
  for (int d = tid; d < 256; d += 256)
    ctxa[d] = a0 * encattr[b * 768 + d] + a1 * encattr[b * 768 + 256 + d] +
              a2 * encattr[b * 768 + 512 + d];
  __syncthreads();
  // ctxhid A row: [q(512), ctx(1024), ctxa(256)]
  for (int j = tid; j < 1792; j += 256) {
    float v = (j < 512) ? q[j] : (j < 1536 ? ctx[j - 512] : ctxa[j - 1536]);
    actx[b * 1792 + j] = (_Float16)v;
  }
  // gp = sigmoid([ctx, ctxa, q, pe] . genp_W + genp_b)
  int tok = (t == 0) ? 1 : trg[b * TT + (t - 1)];
  float gsum = 0.0f;
  for (int j = tid; j < 2048; j += 256) {
    float v;
    if (j < 1024) v = ctx[j];
    else if (j < 1280) v = ctxa[j - 1024];
    else if (j < 1792) v = q[j - 1280];
    else v = embed[(size_t)tok * EE + (j - 1792)];
    gsum += v * genpW[j];
  }
  red[tid] = gsum; __syncthreads();
  for (int o = 128; o > 0; o >>= 1) {
    if (tid < o) red[tid] += red[tid + o];
    __syncthreads();
  }
  if (tid == 0) gp[b] = 1.0f / (1.0f + __expf(-(red[0] + genpb[0])));
}

// ------------------ fused ctxhid GEMM + tanh epilogue ----------------------
// One 32-wave workgroup: ch = tanh([q,ctx,ctxa] @ ctxhid_W^T) (N=512, K=1792);
// writes ch (f16) into the staging columns the vocab GEMM reads as A.
__global__ __launch_bounds__(1024) void dec_ch_fused(
    const _Float16* __restrict__ actx, const _Float16* __restrict__ ctxhidP,
    _Float16* __restrict__ xf16g) {
  __shared__ float chS[8 * 512];
  const int tid = threadIdx.x;
  const int wave = tid >> 5, lane = tid & 31, laneM = lane & 15;
  const int kb = (lane < 16) ? 0 : 8;
  const int m0 = (lane < 16) ? 0 : 8;
  const int tn = wave;                      // 32 N-tiles, one per wave
  const _Float16* Bt = ctxhidP + (size_t)tn * 56 * 512 + lane * 16;
  const _Float16* Arow = actx + (size_t)laneM * 1792 + kb;
  v8f acc = {};
  for (int kt = 0; kt < 56; ++kt) {
    v8h a0 = *(const v8h*)(Arow + kt * 32);
    v8h a1 = *(const v8h*)(Arow + kt * 32 + 16);
    v16h a = __builtin_shufflevector(a0, a1, 0,1,2,3,4,5,6,7,8,9,10,11,12,13,14,15);
    v16h b = *(const v16h*)(Bt + (size_t)kt * 512);
    if (kt + 1 < 56) __builtin_prefetch(Bt + (size_t)(kt + 1) * 512, 0, 3);
    acc = __builtin_amdgcn_wmma_f32_16x16x32_f16(false, a, false, b,
                                                 (short)0, acc, false, false);
  }
  if (m0 < 8) {
#pragma unroll
    for (int r = 0; r < 8; ++r) chS[(m0 + r) * 512 + tn * 16 + laneM] = acc[r];
  }
  __syncthreads();
  for (int idx = tid; idx < BB * HH; idx += 1024) {
    int b = idx >> 9, j = idx & 511;
    xf16g[b * 768 + 256 + j] = (_Float16)tanhf(chS[idx]);
  }
}

// ---------------- vocab softmax + mix + copy scatter (grid = B) -------------
__global__ __launch_bounds__(1024) void dec_out_fused(
    const float* __restrict__ logits, const float* __restrict__ gp,
    const int* __restrict__ src, const float* __restrict__ alphas,
    float* __restrict__ out, int t) {
  __shared__ float red[1024];
  const int b = blockIdx.x, tid = threadIdx.x;
  const float* L = logits + (size_t)b * VV;
  float m = -__builtin_inff();
  for (int v = tid; v < VV; v += 1024) m = fmaxf(m, L[v]);
  red[tid] = m; __syncthreads();
  for (int o = 512; o > 0; o >>= 1) {
    if (tid < o) red[tid] = fmaxf(red[tid], red[tid + o]);
    __syncthreads();
  }
  m = red[0]; __syncthreads();
  float s = 0.0f;
  for (int v = tid; v < VV; v += 1024) s += __expf(L[v] - m);
  red[tid] = s; __syncthreads();
  for (int o = 512; o > 0; o >>= 1) {
    if (tid < o) red[tid] += red[tid + o];
    __syncthreads();
  }
  float g = gp[b];
  float scale = g / red[0];
  float* O = out + ((size_t)b * TT + t) * VV;
  for (int v = tid; v < VV; v += 1024) O[v] = scale * __expf(L[v] - m);
  __threadfence();     // order plain stores before the RMW atomics below
  __syncthreads();
  float cp = 1.0f - g;
  for (int sI = tid; sI < SS; sI += 1024)
    atomicAdd(O + src[b * SS + sI], cp * alphas[b * SS + sI]);
}

// ---------------------------------------------------------------------------
extern "C" void kernel_launch(void* const* d_in, const int* in_sizes, int n_in,
                              void* d_out, int out_size, void* d_ws, size_t ws_size,
                              hipStream_t stream) {
  (void)in_sizes; (void)n_in; (void)out_size; (void)ws_size;
  const int* src      = (const int*)d_in[0];
  const int* src_e    = (const int*)d_in[1];
  const int* trg      = (const int*)d_in[2];
  const int* user     = (const int*)d_in[3];
  const int* product  = (const int*)d_in[4];
  const int* src_mask = (const int*)d_in[5];
  const float* embed    = (const float*)d_in[7];
  const float* encfWih  = (const float*)d_in[8];
  const float* encfWhh  = (const float*)d_in[9];
  const float* encfBih  = (const float*)d_in[10];
  const float* encfBhh  = (const float*)d_in[11];
  const float* encbWih  = (const float*)d_in[12];
  const float* encbWhh  = (const float*)d_in[13];
  const float* encbBih  = (const float*)d_in[14];
  const float* encbBhh  = (const float*)d_in[15];
  const float* uembed   = (const float*)d_in[16];
  const float* pembed   = (const float*)d_in[17];
  const float* mixW     = (const float*)d_in[18];
  const float* mixb     = (const float*)d_in[19];
  const float* encattrW = (const float*)d_in[20];
  const float* encattrb = (const float*)d_in[21];
  const float* initW    = (const float*)d_in[22];
  const float* initb    = (const float*)d_in[23];
  const float* decWih   = (const float*)d_in[24];
  const float* decWhh   = (const float*)d_in[25];
  const float* decBih   = (const float*)d_in[26];
  const float* decBhh   = (const float*)d_in[27];
  const float* attKey   = (const float*)d_in[28];
  const float* attQ     = (const float*)d_in[29];
  const float* attE     = (const float*)d_in[30];
  const float* attrKey  = (const float*)d_in[31];
  const float* attrQ    = (const float*)d_in[32];
  const float* attrE    = (const float*)d_in[33];
  const float* ctxhidW  = (const float*)d_in[34];
  const float* genpW    = (const float*)d_in[35];
  const float* genpb    = (const float*)d_in[36];
  const float* genW     = (const float*)d_in[37];
  float* out = (float*)d_out;

  // ---------------- workspace layout ----------------
  char* p = (char*)d_ws;
  auto bump = [&](size_t bytes) -> char* {
    char* r = p; p += (bytes + 255) & ~(size_t)255; return r;
  };
  _Float16* encfWihP = (_Float16*)bump((size_t)1536 * 256 * 2);
  _Float16* encfWhhP = (_Float16*)bump((size_t)1536 * 512 * 2);
  _Float16* encbWihP = (_Float16*)bump((size_t)1536 * 256 * 2);
  _Float16* encbWhhP = (_Float16*)bump((size_t)1536 * 512 * 2);
  _Float16* decWihP  = (_Float16*)bump((size_t)1536 * 768 * 2);
  _Float16* decWhhP  = (_Float16*)bump((size_t)1536 * 512 * 2);
  _Float16* attKeyP  = (_Float16*)bump((size_t)512 * 1024 * 2);
  _Float16* ctxhidP  = (_Float16*)bump((size_t)512 * 1792 * 2);
  _Float16* genWP    = (_Float16*)bump((size_t)VV * 512 * 2);
  _Float16* xsf16    = (_Float16*)bump((size_t)SS * BB * EE * 2);
  float* gi_all_f = (float*)bump((size_t)SS * BB * 1536 * 4);
  float* gi_all_b = (float*)bump((size_t)SS * BB * 1536 * 4);
  float* ehid32   = (float*)bump((size_t)BB * SS * 1024 * 4);
  _Float16* ehid16 = (_Float16*)bump((size_t)BB * SS * 1024 * 2);
  float* encfin   = (float*)bump((size_t)BB * 1024 * 4);
  float* projkey  = (float*)bump((size_t)BB * SS * 512 * 4);
  float* encattr  = (float*)bump((size_t)BB * 768 * 4);
  float* pka      = (float*)bump((size_t)BB * 3 * 512 * 4);
  float* h        = (float*)bump((size_t)BB * 512 * 4);
  _Float16* xf16  = (_Float16*)bump((size_t)16 * 768 * 2);
  _Float16* actx  = (_Float16*)bump((size_t)16 * 1792 * 2);
  float* logits = (float*)bump((size_t)16 * VV * 4);
  float* alphas = (float*)bump((size_t)BB * SS * 4);
  float* gp     = (float*)bump((size_t)BB * 4);

  auto pack = [&](const float* W, _Float16* P, int N, int K) {
    size_t tot = (size_t)N * K;
    pack_b<<<dim3((unsigned)((tot + 255) / 256)), dim3(256), 0, stream>>>(W, P, N, K);
  };
  pack(encfWih, encfWihP, 1536, 256);
  pack(encfWhh, encfWhhP, 1536, 512);
  pack(encbWih, encbWihP, 1536, 256);
  pack(encbWhh, encbWhhP, 1536, 512);
  pack(decWih, decWihP, 1536, 768);
  pack(decWhh, decWhhP, 1536, 512);
  pack(attKey, attKeyP, 512, 1024);
  pack(ctxhidW, ctxhidP, 512, 1792);
  pack(genW, genWP, VV, 512);

  // source embedding (f16, rows = t*8 + b)
  embed_src_k<<<dim3((SS * BB * EE + 255) / 256), dim3(256), 0, stream>>>(
      embed, src_e, xsf16);

  // batched x@Wih for both encoder directions (M=3200, N=1536, K=256)
  gemm_wmma<<<dim3(1536 / 64, SS * BB / 16), dim3(128), 0, stream>>>(
      xsf16, 256, encfWihP, gi_all_f, 1536, 1536, 256);
  gemm_wmma<<<dim3(1536 / 64, SS * BB / 16), dim3(128), 0, stream>>>(
      xsf16, 256, encbWihP, gi_all_b, 1536, 1536, 256);

  // sequential encoder scans: fwd + bwd run as two concurrent workgroups
  size_t scan_lds = (size_t)16 * 512 * 2 + 8 * 1536 * 4 + 8 * 512 * 4; // 80 KB
  enc_scan<<<dim3(2), dim3(1024), scan_lds, stream>>>(
      gi_all_f, gi_all_b, encfWhhP, encbWhhP,
      encfBih, encfBhh, encbBih, encbBhh, ehid32, ehid16, encfin);

  // proj_key = enc_hidden @ att_key^T  (M=3200, N=512, K=1024)
  gemm_wmma<<<dim3(512 / 64, BB * SS / 16), dim3(128), 0, stream>>>(
      ehid16, 1024, attKeyP, projkey, 512, 512, 1024);

  // attribute path + decoder init state
  init_attrs<<<dim3(BB), dim3(256), 0, stream>>>(
      uembed, pembed, user, product, mixW, mixb, encattrW, encattrb,
      initW, initb, attrKey, encfin, encattr, pka, h, xf16);
  zero_pads<<<dim3((8 * 1792 + 255) / 256), dim3(256), 0, stream>>>(xf16, actx);

  // ---------------- decoder: 100 sequential steps, 5 launches each ---------
  size_t gru_lds = (size_t)16 * 768 * 2 + 16 * 512 * 2 +
                   (size_t)8 * 1536 * 4 * 2;                 // 136 KB
  for (int t = 0; t < TT; ++t) {
    dec_fused_gru<<<dim3(1), dim3(1024), gru_lds, stream>>>(
        embed, trg, xf16, decWihP, decWhhP, decBih, decBhh, h, t);
    dec_attn<<<dim3(BB), dim3(256), 0, stream>>>(
        h, projkey, ehid32, attQ, attE, attrQ, attrE, pka, encattr,
        src_mask, genpW, genpb, embed, trg, alphas, actx, gp, t);
    dec_ch_fused<<<dim3(1), dim3(1024), 0, stream>>>(actx, ctxhidP, xf16);
    // logits = ch @ gen_W^T  (N=32000, K=512); A aliases ch slice of xf16
    gemm_wmma<<<dim3(VV / 64, 1), dim3(128), 0, stream>>>(
        xf16 + 256, 768, genWP, logits, VV, VV, 512);
    dec_out_fused<<<dim3(BB), dim3(1024), 0, stream>>>(
        logits, gp, src, alphas, out, t);
  }
}
